// CostVolume_49959059587715
// MI455X (gfx1250) — matrix operand order repaired and checked by reading, not compiled, not run on hardware
//
#include <hip/hip_runtime.h>
#include <hip/hip_bf16.h>
#include <math.h>

typedef __attribute__((ext_vector_type(16))) _Float16 v16h;
typedef __attribute__((ext_vector_type(8)))  _Float16 v8h;
typedef __attribute__((ext_vector_type(8)))  float    v8f;

#define TPB   256
#define NWAVE 8
#define SLEN  8192

// ---------------- LDS layout (bytes) ----------------
#define OFF_WG1   0        // w_mlp1_0[:,0:10]  -> [64][16] f16 (zero pad)
#define OFF_WX1   2048     // w_xyz1            -> [64][16]
#define OFF_WX2   4096     // w_xyz2            -> [64][16]
#define OFF_WF1   6144     // w_mlp1_0[:,10:74] -> [64][64]
#define OFF_WF2   14336    // w_mlp1_0[:,74:138]-> [64][64]
#define OFF_W11   22528    // w_mlp1_1          -> [64][64]
#define OFF_W20A  30720    // w_mlp2_0[:,0:64]
#define OFF_W20B  38912    // w_mlp2_0[:,64:128]
#define OFF_W21   47104    // w_mlp2_1
#define OFF_W30A  55296    // w_mlp3_0[:,0:64]
#define OFF_W30N  63488    // w_mlp3_0[:,64:128]
#define OFF_W30G  71680    // w_mlp3_0[:,128:192]
#define OFF_W31   79872    // w_mlp3_1
#define OFF_BIAS  88064    // 9 x 64 f32: b11,b20,b21,bx1,bx2,b31,b1_0,zeros,b3_0
#define OFF_X1    90624    // [3][32] f32
#define OFF_X2    91136
#define OFF_F1T   91648    // [32][64] f16 (K-contiguous, B-operand layout)
#define OFF_F2T   95744
#define OFF_PRE1  99840    // [64][32] f32  (Wf1@F1 + b1_0)
#define OFF_PRE2  108032   // [64][32] f32  (Wf2@F2)
#define OFF_PREN  116224   // [64][32] f32  (W30n@F1 + b3_0)
#define OFF_PREG  124416   // [64][32] f32  (W30g@pi_agg)
#define OFF_PAGH  132608   // pi_agg [32][64] f16 (B-operand layout)
#define OFF_PAGF  136704   // pi_agg [64][32] f32 (gather layout)
#define OFF_G     144896   // geo [1024][16] f16 (also f32 feature staging, 16KB)
#define OFF_SCR   177664   // per-wave 8KB scratch: P[32][64] | E[16][64] | H[16][64]
#define SMEM_BYTES 243200

// ---------------- fragment helpers (layouts per ISA 7.12.2, wave32) ----------
__device__ inline v8h ld8h(const _Float16* p) { return *(const v8h*)p; }

__device__ inline v16h cat8(v8h lo, v8h hi) {
  return __builtin_shufflevector(lo, hi, 0,1,2,3,4,5,6,7,8,9,10,11,12,13,14,15);
}

__device__ inline v8f wmma_f16(v16h a, v16h b, v8f c) {
  // D = A(16x32 f16) * B(32x16 f16) + C(16x16 f32)
  return __builtin_amdgcn_wmma_f32_16x16x32_f16(false, a, false, b, (short)0, c,
                                                false, false);
}

// A fragment: weight stored [64][64] f16 row-major (K contiguous).
// lane l holds row m=(l&15)+16*mt; element e <-> K = kc*32 + (e>>3)*16 + (l>>4)*8 + (e&7)
__device__ inline v16h afrag64(const _Float16* W, int mt, int kc, int lane) {
  const _Float16* p = W + (size_t)(((lane & 15) + mt * 16) * 64 + kc * 32 + ((lane >> 4) << 3));
  return cat8(ld8h(p), ld8h(p + 16));
}

// A fragment: weight stored [64][16] f16 (K=16 real, upper half of K=32 zero in regs)
__device__ inline v16h afrag16(const _Float16* W, int mt, int lane) {
  const _Float16* p = W + (size_t)(((lane & 15) + mt * 16) * 16 + ((lane >> 4) << 3));
  v8h z = {};
  return cat8(ld8h(p), z);
}

// B fragment: activation stored [n][64] f16 (K contiguous per column n).
// lane l holds column n=rb+(l&15); element e <-> K = kc*32 + (e>>3)*16 + (l>>4)*8 + (e&7)
__device__ inline v16h bfrag64(const _Float16* P, int rb, int kc, int lane) {
  const _Float16* p = P + (size_t)((rb + (lane & 15)) * 64 + kc * 32 + ((lane >> 4) << 3));
  return cat8(ld8h(p), ld8h(p + 16));
}

// B fragment from geo matrix [1024][16] f16; K=16 real, K=16..31 -> zero lanes
__device__ inline v16h bfragG(const _Float16* Gp, int n0, int lane) {
  v8h z = {};
  const _Float16* p = Gp + (size_t)((n0 + (lane & 15)) * 16 + ((lane >> 4) << 3));
  return cat8(ld8h(p), z);
}

// store C/D accumulator tile (16x16 f32) to [n][64] f16 at column base cb
__device__ inline void store_tile(_Float16* P, int rb, int cb, v8f acc, int lane) {
  _Float16* p = P + (size_t)((rb + (lane & 15)) * 64 + cb + ((lane >> 4) << 3));
  v8h v;
#pragma unroll
  for (int r = 0; r < 8; ++r) v[r] = (_Float16)acc[r];
  *(v8h*)p = v;
}

__device__ inline v8f relu8(v8f a) {
#pragma unroll
  for (int r = 0; r < 8; ++r) a[r] = fmaxf(a[r], 0.0f);
  return a;
}

__device__ inline v8f bias_acc(const float* bp, int mt, int lane) {
  const float* p = bp + mt * 16 + ((lane >> 4) << 3);
  v8f a;
#pragma unroll
  for (int r = 0; r < 8; ++r) a[r] = p[r];
  return a;
}

__device__ inline float red_max16(float v) {
#pragma unroll
  for (int d = 1; d < 16; d <<= 1) v = fmaxf(v, __shfl_xor(v, d, 32));
  return v;
}
__device__ inline float red_sum16(float v) {
#pragma unroll
  for (int d = 1; d < 16; d <<= 1) v += __shfl_xor(v, d, 32);
  return v;
}

// global f32 weight -> LDS f16, column window [c0, c0+ncopy), dst [64][dcols], zero pad
__device__ inline void load_w(_Float16* dst, const float* src, int scols, int c0,
                              int dcols, int ncopy, int tid) {
  for (int idx = tid; idx < 64 * dcols; idx += TPB) {
    int r = idx / dcols, c = idx - r * dcols;
    dst[idx] = (_Float16)((c < ncopy) ? src[r * scols + c0 + c] : 0.0f);
  }
}

// CDNA5 async memory->LDS copy (tracked on ASYNCcnt), GVS mode:
// mem = SADDR(64b) + VADDR(32b byte offset); LDS dest byte addr in VDST VGPR.
__device__ inline void async_ld_b128(int lds_byte_addr, int g_byte_off, const void* base) {
  asm volatile("global_load_async_to_lds_b128 %0, %1, %2 offset:0"
               :: "v"(lds_byte_addr), "v"(g_byte_off), "s"(base)
               : "memory");
}
__device__ inline void wait_async0() {
  asm volatile("s_wait_asynccnt 0x0" ::: "memory");
}

__global__ __launch_bounds__(TPB, 1)
void cost_volume_kernel(
    const float* __restrict__ xyz1, const float* __restrict__ feature1,
    const float* __restrict__ xyz2, const float* __restrict__ feature2,
    const float* __restrict__ w_mlp1_0, const float* __restrict__ b_mlp1_0,
    const float* __restrict__ w_mlp1_1, const float* __restrict__ b_mlp1_1,
    const float* __restrict__ w_xyz1,  const float* __restrict__ b_xyz1,
    const float* __restrict__ w_xyz2,  const float* __restrict__ b_xyz2,
    const float* __restrict__ w_mlp2_0,const float* __restrict__ b_mlp2_0,
    const float* __restrict__ w_mlp2_1,const float* __restrict__ b_mlp2_1,
    const float* __restrict__ w_mlp3_0,const float* __restrict__ b_mlp3_0,
    const float* __restrict__ w_mlp3_1,const float* __restrict__ b_mlp3_1,
    float* __restrict__ out)
{
  extern __shared__ char sm[];
  const int tid  = threadIdx.x;
  const int lane = tid & 31;
  const int wv   = tid >> 5;
  const int bat  = blockIdx.x >> 8;     // batch
  const int blk  = blockIdx.x & 255;    // 32-point block
  const int s0   = blk * 32;

  _Float16* WG1  = (_Float16*)(sm + OFF_WG1);
  _Float16* WX1  = (_Float16*)(sm + OFF_WX1);
  _Float16* WX2  = (_Float16*)(sm + OFF_WX2);
  _Float16* WF1  = (_Float16*)(sm + OFF_WF1);
  _Float16* WF2  = (_Float16*)(sm + OFF_WF2);
  _Float16* W11  = (_Float16*)(sm + OFF_W11);
  _Float16* W20A = (_Float16*)(sm + OFF_W20A);
  _Float16* W20B = (_Float16*)(sm + OFF_W20B);
  _Float16* W21  = (_Float16*)(sm + OFF_W21);
  _Float16* W30A = (_Float16*)(sm + OFF_W30A);
  _Float16* W30N = (_Float16*)(sm + OFF_W30N);
  _Float16* W30G = (_Float16*)(sm + OFF_W30G);
  _Float16* W31  = (_Float16*)(sm + OFF_W31);
  float*    BIASL= (float*)(sm + OFF_BIAS);
  float*    X1   = (float*)(sm + OFF_X1);
  float*    X2   = (float*)(sm + OFF_X2);
  _Float16* F1T  = (_Float16*)(sm + OFF_F1T);
  _Float16* F2T  = (_Float16*)(sm + OFF_F2T);
  float*    PRE1 = (float*)(sm + OFF_PRE1);
  float*    PRE2 = (float*)(sm + OFF_PRE2);
  float*    PREN = (float*)(sm + OFF_PREN);
  float*    PREG = (float*)(sm + OFF_PREG);
  _Float16* PAGH = (_Float16*)(sm + OFF_PAGH);
  float*    PAGF = (float*)(sm + OFF_PAGF);
  _Float16* G    = (_Float16*)(sm + OFF_G);
  _Float16* scrP = (_Float16*)(sm + OFF_SCR + wv * 8192);  // [32][64]
  _Float16* scrE = scrP + 2048;                            // [16][64]
  _Float16* scrH = scrP + 3072;                            // [16][64]

  __builtin_prefetch(w_mlp1_0, 0, 3);   // global_prefetch_b8 of hot weights

  // ---------------- load phase ----------------
  // Async-copy both 64x32 f32 feature blocks into the (currently free) G region:
  // staging layout flat idx = c*32+ii, F1 at G+0, F2 at G+8KB. 16B chunks.
  for (int c4 = tid; c4 < 512; c4 += TPB) {
    int goff = (int)(((((size_t)bat * 64 + (c4 >> 3)) * SLEN) + s0 + ((c4 & 7) << 2)) * 4);
    async_ld_b128(OFF_G + c4 * 16,        goff, (const void*)feature1);
    async_ld_b128(OFF_G + 8192 + c4 * 16, goff, (const void*)feature2);
  }
  if (tid < 96) {
    int d = tid / 32, ii = tid % 32;
    X1[tid] = xyz1[((size_t)bat * 3 + d) * SLEN + s0 + ii];
    X2[tid] = xyz2[((size_t)bat * 3 + d) * SLEN + s0 + ii];
  }
  load_w(WG1,  w_mlp1_0, 138,   0, 16, 10, tid);
  load_w(WF1,  w_mlp1_0, 138,  10, 64, 64, tid);
  load_w(WF2,  w_mlp1_0, 138,  74, 64, 64, tid);
  load_w(W11,  w_mlp1_1,  64,   0, 64, 64, tid);
  load_w(WX1,  w_xyz1,    10,   0, 16, 10, tid);
  load_w(WX2,  w_xyz2,    10,   0, 16, 10, tid);
  load_w(W20A, w_mlp2_0, 128,   0, 64, 64, tid);
  load_w(W20B, w_mlp2_0, 128,  64, 64, 64, tid);
  load_w(W21,  w_mlp2_1,  64,   0, 64, 64, tid);
  load_w(W30A, w_mlp3_0, 192,   0, 64, 64, tid);
  load_w(W30N, w_mlp3_0, 192,  64, 64, 64, tid);
  load_w(W30G, w_mlp3_0, 192, 128, 64, 64, tid);
  load_w(W31,  w_mlp3_1,  64,   0, 64, 64, tid);
  if (tid < 64) {
    BIASL[      tid] = b_mlp1_1[tid];
    BIASL[ 64 + tid] = b_mlp2_0[tid];
    BIASL[128 + tid] = b_mlp2_1[tid];
    BIASL[192 + tid] = b_xyz1[tid];
    BIASL[256 + tid] = b_xyz2[tid];
    BIASL[320 + tid] = b_mlp3_1[tid];
    BIASL[384 + tid] = b_mlp1_0[tid];   // pre-GEMM g==0 bias
    BIASL[448 + tid] = 0.0f;            // pre-GEMM g==1 bias (none)
    BIASL[512 + tid] = b_mlp3_0[tid];   // pre-GEMM g==2 bias
  }
  wait_async0();            // drain ASYNCcnt before anyone reads the staging
  __syncthreads();

  // transpose-convert staged f32 features -> [i][c] f16 B-operand layout
  {
    const float* stage = (const float*)(sm + OFF_G);
    for (int idx = tid; idx < 2048; idx += TPB) {
      int c = idx >> 5, ii = idx & 31;
      F1T[ii * 64 + c] = (_Float16)stage[idx];
      F2T[ii * 64 + c] = (_Float16)stage[2048 + idx];
    }
  }
  __syncthreads();

  // ---------------- stage-1 geometry matrix: columns n = i*32 + k --------------
  for (int idx = tid; idx < 1024; idx += TPB) {
    int i = idx >> 5, k = idx & 31;
    float px = X1[i], py = X1[32 + i], pz = X1[64 + i];
    float qx = X2[k], qy = X2[32 + k], qz = X2[64 + k];
    float dx = qx - px, dy = qy - py, dz = qz - pz;
    float eu = sqrtf(dx * dx + dy * dy + dz * dz + 1e-20f);
    v8h lo = {}, hi = {};
    lo[0] = (_Float16)px; lo[1] = (_Float16)py; lo[2] = (_Float16)pz;
    lo[3] = (_Float16)qx; lo[4] = (_Float16)qy; lo[5] = (_Float16)qz;
    lo[6] = (_Float16)dx; lo[7] = (_Float16)dy;
    hi[0] = (_Float16)dz; hi[1] = (_Float16)eu;
    *(v8h*)(G + idx * 16) = lo;
    *(v8h*)(G + idx * 16 + 8) = hi;
  }

  // ---------------- small pre-GEMMs (64x32, K=64): pre1, pre2, pre_n -----------
  // bias rides in the WMMA C operand -> epilogue is pure stores
  for (int t = wv; t < 24; t += NWAVE) {
    int g = t >> 3, mt = (t >> 1) & 3, nt = t & 1;
    const _Float16* A  = (g == 0) ? WF1 : (g == 1) ? WF2 : W30N;
    const _Float16* Bm = (g == 1) ? F2T : F1T;
    float* OUT = (g == 0) ? PRE1 : (g == 1) ? PRE2 : PREN;
    v8f acc = bias_acc(BIASL + 384 + g * 64, mt, lane);
    acc = wmma_f16(afrag64(A, mt, 0, lane), bfrag64(Bm, nt * 16, 0, lane), acc);
    acc = wmma_f16(afrag64(A, mt, 1, lane), bfrag64(Bm, nt * 16, 1, lane), acc);
    int c0 = mt * 16 + ((lane >> 4) << 3);
    int sp = nt * 16 + (lane & 15);
#pragma unroll
    for (int r = 0; r < 8; ++r)
      OUT[(c0 + r) * 32 + sp] = acc[r];
  }
  __syncthreads();

  // ---------------- stage-1 pipeline: each wave owns i = wv*4 .. wv*4+3 --------
  for (int p = 0; p < 4; ++p) {
    const int i = wv * 4 + p;
    float lg[2][4][8];
#pragma unroll
    for (int t = 0; t < 2; ++t) {
      const int n0 = i * 32 + t * 16;
      // h1 = relu(Wg1@geo + pre1[.,i] + pre2[.,k])  (b1_0 folded into pre1)
#pragma unroll
      for (int mt = 0; mt < 4; ++mt) {
        v8f acc = {};
        acc = wmma_f16(afrag16(WG1, mt, lane), bfragG(G, n0, lane), acc);
        int c0 = mt * 16 + ((lane >> 4) << 3);
        int kk = t * 16 + (lane & 15);
#pragma unroll
        for (int r = 0; r < 8; ++r)
          acc[r] = fmaxf(acc[r] + PRE1[(c0 + r) * 32 + i] + PRE2[(c0 + r) * 32 + kk], 0.0f);
        store_tile(scrH, 0, mt * 16, acc, lane);
      }
      // pi_new = relu(W11@h1 + b1_1)   (kept in pair scratch rows t*16..)
      {
        v16h h0 = bfrag64(scrH, 0, 0, lane), h1 = bfrag64(scrH, 0, 1, lane);
#pragma unroll
        for (int mt = 0; mt < 4; ++mt) {
          v8f acc = bias_acc(BIASL, mt, lane);
          acc = wmma_f16(afrag64(W11, mt, 0, lane), h0, acc);
          acc = wmma_f16(afrag64(W11, mt, 1, lane), h1, acc);
          acc = relu8(acc);
          store_tile(scrP, t * 16, mt * 16, acc, lane);
        }
      }
      // pi_enc = relu(Wx1@geo + bx1)
#pragma unroll
      for (int mt = 0; mt < 4; ++mt) {
        v8f acc = bias_acc(BIASL + 192, mt, lane);
        acc = wmma_f16(afrag16(WX1, mt, lane), bfragG(G, n0, lane), acc);
        acc = relu8(acc);
        store_tile(scrE, 0, mt * 16, acc, lane);
      }
      // h2 = relu(W20a@pi_enc + W20b@pi_new + b2_0)
      {
        v16h e0 = bfrag64(scrE, 0, 0, lane), e1 = bfrag64(scrE, 0, 1, lane);
        v16h p0 = bfrag64(scrP, t * 16, 0, lane), p1 = bfrag64(scrP, t * 16, 1, lane);
#pragma unroll
        for (int mt = 0; mt < 4; ++mt) {
          v8f acc = bias_acc(BIASL + 64, mt, lane);
          acc = wmma_f16(afrag64(W20A, mt, 0, lane), e0, acc);
          acc = wmma_f16(afrag64(W20A, mt, 1, lane), e1, acc);
          acc = wmma_f16(afrag64(W20B, mt, 0, lane), p0, acc);
          acc = wmma_f16(afrag64(W20B, mt, 1, lane), p1, acc);
          acc = relu8(acc);
          store_tile(scrH, 0, mt * 16, acc, lane);
        }
      }
      // logits = W21@h2 + b2_1
      {
        v16h h0 = bfrag64(scrH, 0, 0, lane), h1 = bfrag64(scrH, 0, 1, lane);
#pragma unroll
        for (int mt = 0; mt < 4; ++mt) {
          v8f acc = bias_acc(BIASL + 128, mt, lane);
          acc = wmma_f16(afrag64(W21, mt, 0, lane), h0, acc);
          acc = wmma_f16(afrag64(W21, mt, 1, lane), h1, acc);
#pragma unroll
          for (int r = 0; r < 8; ++r) lg[t][mt][r] = acc[r];
        }
      }
    }
    // softmax over k (32 = 2 tiles x 16 lanes) + weighted pooling of pi_new
#pragma unroll
    for (int mt = 0; mt < 4; ++mt) {
      int c0 = mt * 16 + ((lane >> 4) << 3);
#pragma unroll
      for (int r = 0; r < 8; ++r) {
        float a = lg[0][mt][r], bb = lg[1][mt][r];
        float mx  = red_max16(fmaxf(a, bb));
        float ea  = __expf(a - mx), eb = __expf(bb - mx);
        float den = red_sum16(ea + eb);
        float pa  = (float)scrP[(lane & 15) * 64 + c0 + r];
        float pb  = (float)scrP[(16 + (lane & 15)) * 64 + c0 + r];
        float num = red_sum16(ea * pa + eb * pb);
        float agg = num / den;
        if ((lane & 15) == 0) {
          PAGH[i * 64 + c0 + r] = (_Float16)agg;
          PAGF[(c0 + r) * 32 + i] = agg;
        }
      }
    }
  }
  __syncthreads();

  // ---------------- stage-2 geometry: columns n = i*32 + j (j=31 is dummy) -----
  for (int idx = tid; idx < 1024; idx += TPB) {
    int i2 = idx >> 5, j = idx & 31;
    v8h lo = {}, hi = {};
    if (j < 31) {
      int q = j + ((j >= i2) ? 1 : 0);
      float px = X1[i2], py = X1[32 + i2], pz = X1[64 + i2];
      float qx = X1[q],  qy = X1[32 + q],  qz = X1[64 + q];
      float dx = qx - px, dy = qy - py, dz = qz - pz;
      float eu = sqrtf(dx * dx + dy * dy + dz * dz + 1e-20f);
      lo[0] = (_Float16)px; lo[1] = (_Float16)py; lo[2] = (_Float16)pz;
      lo[3] = (_Float16)qx; lo[4] = (_Float16)qy; lo[5] = (_Float16)qz;
      lo[6] = (_Float16)dx; lo[7] = (_Float16)dy;
      hi[0] = (_Float16)dz; hi[1] = (_Float16)eu;
    }
    *(v8h*)(G + idx * 16) = lo;
    *(v8h*)(G + idx * 16 + 8) = hi;
  }
  // pre_g = W30g @ pi_agg  (one 16x16 tile per wave)
  {
    int mt = wv >> 1, nt = wv & 1;
    v8f acc = {};
    acc = wmma_f16(afrag64(W30G, mt, 0, lane), bfrag64(PAGH, nt * 16, 0, lane), acc);
    acc = wmma_f16(afrag64(W30G, mt, 1, lane), bfrag64(PAGH, nt * 16, 1, lane), acc);
    int c0 = mt * 16 + ((lane >> 4) << 3);
    int sp = nt * 16 + (lane & 15);
#pragma unroll
    for (int r = 0; r < 8; ++r) PREG[(c0 + r) * 32 + sp] = acc[r];
  }
  __syncthreads();

  // ---------------- stage-2 pipeline ----------------
  for (int p = 0; p < 4; ++p) {
    const int i = wv * 4 + p;
    float lg[2][4][8];
#pragma unroll
    for (int t = 0; t < 2; ++t) {
      const int n0 = i * 32 + t * 16;
      // pc_enc = relu(Wx2@geo2 + bx2)
#pragma unroll
      for (int mt = 0; mt < 4; ++mt) {
        v8f acc = bias_acc(BIASL + 256, mt, lane);
        acc = wmma_f16(afrag16(WX2, mt, lane), bfragG(G, n0, lane), acc);
        acc = relu8(acc);
        store_tile(scrE, 0, mt * 16, acc, lane);
      }
      // h3 = relu(W30a@pc_enc + pre_n[.,i] + pre_g[.,q])  (b3_0 folded into pre_n)
      {
        v16h e0 = bfrag64(scrE, 0, 0, lane), e1 = bfrag64(scrE, 0, 1, lane);
        int j = t * 16 + (lane & 15);
        int q = (j < 31) ? (j + ((j >= i) ? 1 : 0)) : 0;
#pragma unroll
        for (int mt = 0; mt < 4; ++mt) {
          v8f acc = {};
          acc = wmma_f16(afrag64(W30A, mt, 0, lane), e0, acc);
          acc = wmma_f16(afrag64(W30A, mt, 1, lane), e1, acc);
          int c0 = mt * 16 + ((lane >> 4) << 3);
#pragma unroll
          for (int r = 0; r < 8; ++r)
            acc[r] = fmaxf(acc[r] + PREN[(c0 + r) * 32 + i] + PREG[(c0 + r) * 32 + q], 0.0f);
          store_tile(scrH, 0, mt * 16, acc, lane);
        }
      }
      // logits = W31@h3 + b3_1
      {
        v16h h0 = bfrag64(scrH, 0, 0, lane), h1 = bfrag64(scrH, 0, 1, lane);
#pragma unroll
        for (int mt = 0; mt < 4; ++mt) {
          v8f acc = bias_acc(BIASL + 320, mt, lane);
          acc = wmma_f16(afrag64(W31, mt, 0, lane), h0, acc);
          acc = wmma_f16(afrag64(W31, mt, 1, lane), h1, acc);
#pragma unroll
          for (int r = 0; r < 8; ++r) lg[t][mt][r] = acc[r];
        }
      }
    }
    // softmax over j (mask j=31) + weighted pooling of gathered pi_agg -> output
    int ja = lane & 15;
    int qa = ja + ((ja >= i) ? 1 : 0);
    int jb = 16 + (lane & 15);
    bool mB = (jb == 31);
    int qb = mB ? 0 : (jb + ((jb >= i) ? 1 : 0));
#pragma unroll
    for (int mt = 0; mt < 4; ++mt) {
      int c0 = mt * 16 + ((lane >> 4) << 3);
#pragma unroll
      for (int r = 0; r < 8; ++r) {
        float a  = lg[0][mt][r];
        float bb = mB ? -3.0e38f : lg[1][mt][r];
        float mx  = red_max16(fmaxf(a, bb));
        float ea  = __expf(a - mx);
        float eb  = mB ? 0.0f : __expf(bb - mx);
        float den = red_sum16(ea + eb);
        float pa  = PAGF[(c0 + r) * 32 + qa];
        float pb  = mB ? 0.0f : PAGF[(c0 + r) * 32 + qb];
        float num = red_sum16(ea * pa + eb * pb);
        float agg = num / den;
        if ((lane & 15) == 0)
          out[((size_t)bat * 64 + c0 + r) * SLEN + s0 + i] = agg;
      }
    }
  }
}

extern "C" void kernel_launch(void* const* d_in, const int* in_sizes, int n_in,
                              void* d_out, int out_size, void* d_ws, size_t ws_size,
                              hipStream_t stream) {
  (void)in_sizes; (void)n_in; (void)out_size; (void)d_ws; (void)ws_size;
  hipFuncSetAttribute(reinterpret_cast<const void*>(cost_volume_kernel),
                      hipFuncAttributeMaxDynamicSharedMemorySize, SMEM_BYTES);
  cost_volume_kernel<<<dim3(512), dim3(TPB), SMEM_BYTES, stream>>>(
      (const float*)d_in[0],  (const float*)d_in[1],
      (const float*)d_in[2],  (const float*)d_in[3],
      (const float*)d_in[4],  (const float*)d_in[5],
      (const float*)d_in[6],  (const float*)d_in[7],
      (const float*)d_in[8],  (const float*)d_in[9],
      (const float*)d_in[10], (const float*)d_in[11],
      (const float*)d_in[12], (const float*)d_in[13],
      (const float*)d_in[14], (const float*)d_in[15],
      (const float*)d_in[16], (const float*)d_in[17],
      (const float*)d_in[18], (const float*)d_in[19],
      (float*)d_out);
}
